// GAT_MN_67980742361114
// MI455X (gfx1250) — matrix operand order repaired and checked by reading, not compiled, not run on hardware
//
#include <hip/hip_runtime.h>
#include <hip/hip_bf16.h>
#include <math.h>

// ---------------------------------------------------------------------------
// GAT (2 layers, 4 heads x 32 dim) + mean pooling + classifier for gfx1250.
//   layer1 GEMM: [N,4]x[4,128]     -> v_wmma_f32_16x16x4_f32  (f32, K=4)
//   layer2 GEMM: [N,128]x[128,128] -> v_wmma_f32_16x16x32_f16 (f16 in, f32 acc)
//     - A tile (16x128, 4KB) + full W2 (32KB) staged in LDS per block via
//       gfx1250 async global->LDS (ASYNCcnt) with 8 waves sharing the A tile.
// Edge softmax/aggregation is scatter-bound; out+ft (~51MB) is L2-resident
// (192MB L2) so float atomics resolve in L2 -> left as atomics.
// ---------------------------------------------------------------------------

typedef _Float16 f16;
typedef __attribute__((ext_vector_type(16))) _Float16 v16h;
typedef __attribute__((ext_vector_type(8)))  _Float16 v8h;
typedef __attribute__((ext_vector_type(8)))  float    v8f;
typedef __attribute__((ext_vector_type(2)))  float    v2f;

#define NN      50000
#define NE      800000
#define NG      128
#define HEADS   4
#define HDIM    32
#define FDIM    128      // HEADS*HDIM
#define NCLS    10
#define LEAKY   0.2f
#define TILES_M 3125     // NN/16 exactly

// Use gfx1250 async global->LDS staging (flip to 0 if assembler rejects).
#define USE_ASYNC_LDS 1

// 16-byte global -> LDS copy.
__device__ __forceinline__ void gat_g2l_16B(const void* gptr, void* lptr) {
#if USE_ASYNC_LDS
    // VDST = LDS byte offset (low 32 bits of the shared-aperture address),
    // VADDR = 64-bit global address.  Tracked by ASYNCcnt.
    unsigned lds = (unsigned)(unsigned long long)lptr;
    unsigned long long g = (unsigned long long)gptr;
    asm volatile("global_load_async_to_lds_b128 %0, %1, off"
                 :: "v"(lds), "v"(g) : "memory");
#else
    *(v8h*)lptr = *(const v8h*)gptr;
#endif
}

__device__ __forceinline__ void gat_wait_async() {
#if USE_ASYNC_LDS
#if __has_builtin(__builtin_amdgcn_s_wait_asynccnt)
    __builtin_amdgcn_s_wait_asynccnt(0);
#else
    asm volatile("s_wait_asynccnt 0x0" ::: "memory");
#endif
#endif
}

// ----------------------------- utility kernels -----------------------------

__global__ void gat_fill_f32(float* __restrict__ p, long long n, float v) {
    long long i = (long long)blockIdx.x * blockDim.x + threadIdx.x;
    if (i < n) p[i] = v;
}

__global__ void gat_degree(const int* __restrict__ dst, float* __restrict__ deg) {
    int e = blockIdx.x * blockDim.x + threadIdx.x;
    if (e < NE) atomicAdd(&deg[dst[e]], 1.0f);
}

__global__ void gat_feat(const float* __restrict__ deg, float* __restrict__ h0) {
    int n = blockIdx.x * blockDim.x + threadIdx.x;
    if (n < NN) {
        float d = deg[n];
        h0[n * 4 + 0] = d;
        h0[n * 4 + 1] = (d - 3.0f > 0.0f) ? 1.0f : 0.0f;
        h0[n * 4 + 2] = 3.0f / d;
        h0[n * 4 + 3] = (d - 4.0f > 0.0f) ? 1.0f : 0.0f;
    }
}

__global__ void gat_relu(float* __restrict__ p, long long n) {
    long long i = (long long)blockIdx.x * blockDim.x + threadIdx.x;
    if (i < n) p[i] = fmaxf(p[i], 0.0f);
}

__global__ void gat_cvt_f16(const float* __restrict__ in, f16* __restrict__ out,
                            long long n) {
    long long i = (long long)blockIdx.x * blockDim.x + threadIdx.x;
    if (i < n) out[i] = (f16)in[i];
}

// ------------------------- WMMA GEMM, K=4, f32 -----------------------------
// C[N,128] = A[N,4] x B[4,128].  One wave per 16x16 tile, single K=4 WMMA.
__global__ void gat_gemm_k4_wmma(const float* __restrict__ A,
                                 const float* __restrict__ B,
                                 float* __restrict__ C) {
    int tm = blockIdx.x;          // 0..3124
    int tn = blockIdx.y;          // 0..7
    int lane = threadIdx.x & 31;
    int r  = lane & 15;
    int hf = lane >> 4;
    int node = tm * 16 + r;
    int col  = tn * 16 + r;

    v2f a;
    a.x = A[(size_t)node * 4 + 2 * hf + 0];
    a.y = A[(size_t)node * 4 + 2 * hf + 1];
    v2f b;
    b.x = B[(size_t)(2 * hf + 0) * FDIM + col];
    b.y = B[(size_t)(2 * hf + 1) * FDIM + col];

    v8f c = {};
    c = __builtin_amdgcn_wmma_f32_16x16x4_f32(false, a, false, b,
                                              (short)0, c, false, false);
#pragma unroll
    for (int i = 0; i < 8; ++i)
        C[(size_t)(tm * 16 + 8 * hf + i) * FDIM + col] = c[i];
}

// ------------------------ WMMA GEMM, K=128, f16 ----------------------------
// C[N,128] = A[N,128] x B[128,128].
// One block (8 wave32) per 16-row stripe; wave w owns N-tile tn=w.
// A tile (16x128 halfs, 4KB) and full B (128x128 halfs, 32KB) are staged in
// LDS with async global->LDS b128 copies, then fragments are built from LDS:
//   A frag (16x32 f16): lane row r=lane&15, hf=lane>>4:
//     halfs 0..7  -> K = kt*32 + 8*hf + 0..7   (16B LDS load)
//     halfs 8..15 -> K = kt*32 + 16 + 8*hf ... (16B LDS load)
//   B frag (32x16 f16): lane col = 16*w + r; half j -> K = kt*32 + 8*hf + j (j<8), kt*32 + 16 + 8*hf + j-8 (j>=8).
//   C frag (16x16 f32): col = 16*w + r; rows 8*hf + 0..7.
__global__ void __launch_bounds__(256)
gat_gemm_f16_wmma(const f16* __restrict__ A,
                  const f16* __restrict__ B,
                  float* __restrict__ C) {
    __shared__ __align__(16) f16 Atile[16 * FDIM];     // 4 KB
    __shared__ __align__(16) f16 Btile[FDIM * FDIM];   // 32 KB

    int tm   = blockIdx.x;
    int t    = threadIdx.x;        // 0..255
    int w    = t >> 5;             // wave -> tn (0..7)
    int lane = t & 31;

    // Stage A tile: 256 chunks of 16B, one per thread.
    gat_g2l_16B(A + (size_t)tm * 16 * FDIM + (size_t)t * 8, Atile + t * 8);
    // Stage full W2: 2048 chunks of 16B, 8 per thread.
#pragma unroll
    for (int i = 0; i < 8; ++i) {
        int c = t + i * 256;
        gat_g2l_16B(B + (size_t)c * 8, Btile + (size_t)c * 8);
    }
    gat_wait_async();
    __syncthreads();

    int r  = lane & 15;
    int hf = lane >> 4;
    int col = w * 16 + r;

    v8f acc = {};
#pragma unroll
    for (int kt = 0; kt < 4; ++kt) {
        const v8h* ap = (const v8h*)(Atile + r * FDIM + kt * 32 + 8 * hf);
        v8h lo = ap[0];            // K = kt*32 + 8*hf + 0..7
        v8h hi = ap[2];            // K = kt*32 + 8*hf + 16..23
        v16h a;
#pragma unroll
        for (int j = 0; j < 8; ++j) { a[j] = lo[j]; a[j + 8] = hi[j]; }

        const f16* bp = Btile + (size_t)(kt * 32 + 8 * hf) * FDIM + col;
        v16h b;
#pragma unroll
        for (int j = 0; j < 16; ++j) b[j] = bp[(size_t)(j < 8 ? j : j + 8) * FDIM];

        acc = __builtin_amdgcn_wmma_f32_16x16x32_f16(false, a, false, b,
                                                     (short)0, acc, false, false);
    }
#pragma unroll
    for (int i = 0; i < 8; ++i)
        C[(size_t)(tm * 16 + 8 * hf + i) * FDIM + col] = acc[i];
}

// ------------------------- attention / edge kernels ------------------------

__global__ void gat_attn_coef(const float* __restrict__ ft,
                              const float* __restrict__ al,
                              const float* __restrict__ ar,
                              float* __restrict__ aln,
                              float* __restrict__ arn) {
    int idx = blockIdx.x * blockDim.x + threadIdx.x;   // node*HEADS + h
    if (idx < NN * HEADS) {
        int n = idx / HEADS, h = idx % HEADS;
        const float* f = ft + (size_t)n * FDIM + h * HDIM;
        const float* L = al + h * HDIM;
        const float* R = ar + h * HDIM;
        float sl = 0.0f, sr = 0.0f;
#pragma unroll 8
        for (int d = 0; d < HDIM; ++d) { float v = f[d]; sl += v * L[d]; sr += v * R[d]; }
        aln[idx] = sl;
        arn[idx] = sr;
    }
}

__device__ inline float gatAtomicMaxF(float* addr, float val) {
    if (val >= 0.0f)
        return __int_as_float(atomicMax((int*)addr, __float_as_int(val)));
    else
        return __uint_as_float(atomicMin((unsigned int*)addr, __float_as_uint(val)));
}

__global__ void gat_edge_emax(const int* __restrict__ src,
                              const int* __restrict__ dst,
                              const float* __restrict__ aln,
                              const float* __restrict__ arn,
                              float* __restrict__ ebuf,
                              float* __restrict__ mmax) {
    int e = blockIdx.x * blockDim.x + threadIdx.x;
    if (e < NE) {
        int s = src[e], d = dst[e];
#pragma unroll
        for (int h = 0; h < HEADS; ++h) {
            float x = aln[s * HEADS + h] + arn[d * HEADS + h];
            x = (x > 0.0f) ? x : LEAKY * x;
            ebuf[(size_t)e * HEADS + h] = x;
            gatAtomicMaxF(&mmax[d * HEADS + h], x);
        }
    }
}

__global__ void gat_edge_expsum(const int* __restrict__ dst,
                                const float* __restrict__ ebuf,
                                const float* __restrict__ mmax,
                                float* __restrict__ den) {
    int e = blockIdx.x * blockDim.x + threadIdx.x;
    if (e < NE) {
        int d = dst[e];
#pragma unroll
        for (int h = 0; h < HEADS; ++h)
            atomicAdd(&den[d * HEADS + h],
                      expf(ebuf[(size_t)e * HEADS + h] - mmax[d * HEADS + h]));
    }
}

// 32 lanes per edge: lane = head-dim index, 4 atomics/lane (one per head).
__global__ void gat_edge_agg(const int* __restrict__ src,
                             const int* __restrict__ dst,
                             const float* __restrict__ ebuf,
                             const float* __restrict__ mmax,
                             const float* __restrict__ den,
                             const float* __restrict__ ft,
                             float* __restrict__ out) {
    long long t = (long long)blockIdx.x * blockDim.x + threadIdx.x;
    int e    = (int)(t >> 5);
    int lane = (int)(t & 31);
    if (e < NE) {
        int s = src[e], d = dst[e];
#pragma unroll
        for (int h = 0; h < HEADS; ++h) {
            float alpha = expf(ebuf[(size_t)e * HEADS + h] - mmax[d * HEADS + h])
                          / den[d * HEADS + h];
            int f = h * HDIM + lane;
            atomicAdd(&out[(size_t)d * FDIM + f],
                      ft[(size_t)s * FDIM + f] * alpha);
        }
    }
}

// ------------------------------- pooling -----------------------------------

__global__ void gat_pool_sum(const int* __restrict__ gid,
                             const float* __restrict__ h,
                             float* __restrict__ gsum) {
    long long t = (long long)blockIdx.x * blockDim.x + threadIdx.x;
    int n    = (int)(t >> 5);
    int lane = (int)(t & 31);
    if (n < NN) {
        int g = gid[n];
#pragma unroll
        for (int j = 0; j < 4; ++j) {
            int f = j * 32 + lane;
            atomicAdd(&gsum[(size_t)g * FDIM + f], h[(size_t)n * FDIM + f]);
        }
    }
}

__global__ void gat_pool_cnt(const int* __restrict__ gid, float* __restrict__ cnt) {
    int n = blockIdx.x * blockDim.x + threadIdx.x;
    if (n < NN) atomicAdd(&cnt[gid[n]], 1.0f);
}

__global__ void gat_classify(const float* __restrict__ gsum,
                             const float* __restrict__ cnt,
                             const float* __restrict__ Wc,
                             const float* __restrict__ bc,
                             float* __restrict__ out) {
    int t = blockIdx.x * blockDim.x + threadIdx.x;
    if (t < NG * NCLS) {
        int g = t / NCLS, c = t % NCLS;
        float inv = 1.0f / fmaxf(cnt[g], 1.0f);
        float s = bc[c];
        for (int f = 0; f < FDIM; ++f)
            s += gsum[(size_t)g * FDIM + f] * inv * Wc[f * NCLS + c];
        out[t] = 1.0f / (1.0f + expf(-s));
    }
}

// ------------------------------ launcher -----------------------------------

extern "C" void kernel_launch(void* const* d_in, const int* in_sizes, int n_in,
                              void* d_out, int out_size, void* d_ws, size_t ws_size,
                              hipStream_t stream) {
    (void)in_sizes; (void)n_in; (void)out_size; (void)ws_size;

    const int*   src = (const int*)d_in[0];
    const int*   dst = (const int*)d_in[1];
    const int*   gid = (const int*)d_in[2];
    const float* W1  = (const float*)d_in[3];
    const float* al1 = (const float*)d_in[4];
    const float* ar1 = (const float*)d_in[5];
    const float* W2  = (const float*)d_in[6];
    const float* al2 = (const float*)d_in[7];
    const float* ar2 = (const float*)d_in[8];
    const float* Wc  = (const float*)d_in[9];
    const float* bc  = (const float*)d_in[10];
    float* out = (float*)d_out;

    // Workspace layout (float units), ~81 MB total.
    float* ws = (float*)d_ws;
    const size_t OFF_DEG  = 0;
    const size_t OFF_H0   = OFF_DEG  + NN;               // N*4
    const size_t OFF_FT   = OFF_H0   + (size_t)NN * 4;   // N*128
    const size_t OFF_OUT  = OFF_FT   + (size_t)NN * FDIM;// N*128
    const size_t OFF_AL   = OFF_OUT  + (size_t)NN * FDIM;// N*4
    const size_t OFF_AR   = OFF_AL   + (size_t)NN * HEADS;
    const size_t OFF_M    = OFF_AR   + (size_t)NN * HEADS;
    const size_t OFF_DEN  = OFF_M    + (size_t)NN * HEADS;
    const size_t OFF_E    = OFF_DEN  + (size_t)NN * HEADS; // E*4
    const size_t OFF_H16  = OFF_E    + (size_t)NE * HEADS; // N*128 halfs
    const size_t OFF_W2H  = OFF_H16  + (size_t)NN * (FDIM / 2); // 128*128 halfs
    const size_t OFF_GSUM = OFF_W2H  + (size_t)(FDIM * FDIM) / 2;
    const size_t OFF_GCNT = OFF_GSUM + (size_t)NG * FDIM;

    float* deg  = ws + OFF_DEG;
    float* h0   = ws + OFF_H0;
    float* ftA  = ws + OFF_FT;
    float* outB = ws + OFF_OUT;
    float* aln  = ws + OFF_AL;
    float* arn  = ws + OFF_AR;
    float* mmax = ws + OFF_M;
    float* den  = ws + OFF_DEN;
    float* ebuf = ws + OFF_E;
    f16*   h16  = (f16*)(ws + OFF_H16);
    f16*   W2h  = (f16*)(ws + OFF_W2H);
    float* gsum = ws + OFF_GSUM;
    float* gcnt = ws + OFF_GCNT;

    const int T = 256;
    auto blocks = [](long long n, int t) { return (unsigned)((n + t - 1) / t); };

    // ---- degrees & input features ----
    gat_fill_f32<<<blocks(NN, T), T, 0, stream>>>(deg, NN, 0.0f);
    gat_degree<<<blocks(NE, T), T, 0, stream>>>(dst, deg);
    gat_feat<<<blocks(NN, T), T, 0, stream>>>(deg, h0);

    // ---- layer 1 GEMM: ft = h0 @ W1  (WMMA f32 K=4) ----
    gat_gemm_k4_wmma<<<dim3(TILES_M, FDIM / 16), 32, 0, stream>>>(h0, W1, ftA);

    // ---- layer 1 attention + aggregation ----
    gat_fill_f32<<<blocks((long long)NN * HEADS, T), T, 0, stream>>>(mmax, (long long)NN * HEADS, -INFINITY);
    gat_fill_f32<<<blocks((long long)NN * HEADS, T), T, 0, stream>>>(den, (long long)NN * HEADS, 0.0f);
    gat_fill_f32<<<blocks((long long)NN * FDIM, T), T, 0, stream>>>(outB, (long long)NN * FDIM, 0.0f);
    gat_attn_coef<<<blocks((long long)NN * HEADS, T), T, 0, stream>>>(ftA, al1, ar1, aln, arn);
    gat_edge_emax<<<blocks(NE, T), T, 0, stream>>>(src, dst, aln, arn, ebuf, mmax);
    gat_edge_expsum<<<blocks(NE, T), T, 0, stream>>>(dst, ebuf, mmax, den);
    gat_edge_agg<<<blocks((long long)NE * 32, T), T, 0, stream>>>(src, dst, ebuf, mmax, den, ftA, outB);
    gat_relu<<<blocks((long long)NN * FDIM, T), T, 0, stream>>>(outB, (long long)NN * FDIM);

    // ---- layer 2 GEMM: ft = h1 @ W2  (WMMA f16->f32, K=128, LDS-staged) ----
    gat_cvt_f16<<<blocks((long long)NN * FDIM, T), T, 0, stream>>>(outB, h16, (long long)NN * FDIM);
    gat_cvt_f16<<<blocks((long long)FDIM * FDIM, T), T, 0, stream>>>(W2, W2h, (long long)FDIM * FDIM);
    gat_gemm_f16_wmma<<<dim3(TILES_M), 256, 0, stream>>>(h16, W2h, ftA);

    // ---- layer 2 attention + aggregation ----
    gat_fill_f32<<<blocks((long long)NN * HEADS, T), T, 0, stream>>>(mmax, (long long)NN * HEADS, -INFINITY);
    gat_fill_f32<<<blocks((long long)NN * HEADS, T), T, 0, stream>>>(den, (long long)NN * HEADS, 0.0f);
    gat_fill_f32<<<blocks((long long)NN * FDIM, T), T, 0, stream>>>(outB, (long long)NN * FDIM, 0.0f);
    gat_attn_coef<<<blocks((long long)NN * HEADS, T), T, 0, stream>>>(ftA, al2, ar2, aln, arn);
    gat_edge_emax<<<blocks(NE, T), T, 0, stream>>>(src, dst, aln, arn, ebuf, mmax);
    gat_edge_expsum<<<blocks(NE, T), T, 0, stream>>>(dst, ebuf, mmax, den);
    gat_edge_agg<<<blocks((long long)NE * 32, T), T, 0, stream>>>(src, dst, ebuf, mmax, den, ftA, outB);
    gat_relu<<<blocks((long long)NN * FDIM, T), T, 0, stream>>>(outB, (long long)NN * FDIM);

    // ---- mean pooling + classifier ----
    gat_fill_f32<<<blocks((long long)NG * FDIM + NG, T), T, 0, stream>>>(gsum, (long long)NG * FDIM + NG, 0.0f);
    gat_pool_sum<<<blocks((long long)NN * 32, T), T, 0, stream>>>(gid, outB, gsum);
    gat_pool_cnt<<<blocks(NN, T), T, 0, stream>>>(gid, gcnt);
    gat_classify<<<blocks(NG * NCLS, T), T, 0, stream>>>(gsum, gcnt, Wc, bc, out);
}
